// NTXentLoss_41790031790556
// MI455X (gfx1250) — matrix operand order repaired and checked by reading, not compiled, not run on hardware
//
#include <hip/hip_runtime.h>
#include <hip/hip_bf16.h>

typedef __attribute__((ext_vector_type(16))) _Float16 v16h;
typedef __attribute__((ext_vector_type(8)))  _Float16 v8h;
typedef __attribute__((ext_vector_type(4)))  _Float16 v4h;
typedef __attribute__((ext_vector_type(8)))  float    v8f;
typedef __attribute__((ext_vector_type(4)))  unsigned int u32x4;
typedef __attribute__((ext_vector_type(8)))  int          i32x8;
typedef __attribute__((ext_vector_type(4)))  int          i32x4;

#define TWO_N    8192
#define NHALF    4096
#define DDIM     512
#define INV_T    14.285714285714286f   // 1/0.07, also the hard upper bound on sim
#define LOG2E    1.4426950408889634f
#define LN2      0.6931471805599453f
#define BSTRIDE  520                    // 512 + 8 halves: TDM pad (4dw per 256dw)

// ---------------------------------------------------------------------------
// Kernel 1: row L2-normalize z=[z1;z2], write f16 copy + keep f32 norms.
// ---------------------------------------------------------------------------
__global__ __launch_bounds__(256) void ntx_normalize(const float* __restrict__ z1,
                                                     const float* __restrict__ z2,
                                                     _Float16* __restrict__ znh,
                                                     float* __restrict__ norms) {
    const int row  = blockIdx.x * 8 + (threadIdx.x >> 5);
    const int lane = threadIdx.x & 31;
    const float* src = (row < NHALF) ? (z1 + (size_t)row * DDIM)
                                     : (z2 + (size_t)(row - NHALF) * DDIM);
    float4 vbuf[4];
    float ss = 0.0f;
#pragma unroll
    for (int p = 0; p < 4; ++p) {
        float4 v = ((const float4*)src)[lane + p * 32];
        vbuf[p] = v;
        ss += v.x * v.x + v.y * v.y + v.z * v.z + v.w * v.w;
    }
    ss += __shfl_xor(ss, 1, 32);
    ss += __shfl_xor(ss, 2, 32);
    ss += __shfl_xor(ss, 4, 32);
    ss += __shfl_xor(ss, 8, 32);
    ss += __shfl_xor(ss, 16, 32);
    float nrm = fmaxf(__builtin_amdgcn_sqrtf(ss), 1e-8f);
    float inv = 1.0f / nrm;
    if (lane == 0) norms[row] = nrm;

    _Float16* dst = znh + (size_t)row * DDIM;
#pragma unroll
    for (int p = 0; p < 4; ++p) {
        float4 v = vbuf[p];
        v4h h = { (_Float16)(v.x * inv), (_Float16)(v.y * inv),
                  (_Float16)(v.z * inv), (_Float16)(v.w * inv) };
        ((v4h*)dst)[lane + p * 32] = h;
    }
}

// ---------------------------------------------------------------------------
// Kernel 2: exact f32 positives: pos[i] = pos[i+N] = dot(z1_i,z2_i)/(n_i*n_{i+N}*T)
// ---------------------------------------------------------------------------
__global__ __launch_bounds__(256) void ntx_positives(const float* __restrict__ z1,
                                                     const float* __restrict__ z2,
                                                     const float* __restrict__ norms,
                                                     float* __restrict__ pos) {
    const int i    = blockIdx.x * 8 + (threadIdx.x >> 5);
    const int lane = threadIdx.x & 31;
    const float4* a = (const float4*)(z1 + (size_t)i * DDIM);
    const float4* b = (const float4*)(z2 + (size_t)i * DDIM);
    float dot = 0.0f;
#pragma unroll
    for (int p = 0; p < 4; ++p) {
        float4 x = a[lane + p * 32];
        float4 y = b[lane + p * 32];
        dot += x.x * y.x + x.y * y.y + x.z * y.z + x.w * y.w;
    }
    dot += __shfl_xor(dot, 1, 32);
    dot += __shfl_xor(dot, 2, 32);
    dot += __shfl_xor(dot, 4, 32);
    dot += __shfl_xor(dot, 8, 32);
    dot += __shfl_xor(dot, 16, 32);
    if (lane == 0) {
        float p = dot / (norms[i] * norms[i + NHALF] * 0.07f);
        pos[i] = p;
        pos[i + NHALF] = p;
    }
}

// ---------------------------------------------------------------------------
// TDM descriptor: load one 16-row x 512-half tile (row-major, 1024B rows)
// from global into LDS at byte offset lds_off, inserting 16B of padding after
// every 1024B row (-> 520-half LDS row stride, bank-conflict-free reads).
// D# bitfields per CDNA5 ISA ch.8 (group0: count/lds/global/type,
// group1: data_size=2B, pad 4dw@256dw, tensor 512x16, tile 512x16, stride 512).
// ---------------------------------------------------------------------------
__device__ __forceinline__ void tdm_load_tile(const _Float16* gsrc, unsigned lds_off) {
    unsigned long long ga = (unsigned long long)(uintptr_t)gsrc;
    u32x4 g0;
    g0[0] = 1u;                                            // count=1, no gather
    g0[1] = lds_off;                                       // LDS byte address
    g0[2] = (unsigned)(ga & 0xFFFFFFFFu);                  // global_addr[31:0]
    g0[3] = (unsigned)((ga >> 32) & 0x01FFFFFFu)           // global_addr[56:32]
          | (2u << 30);                                    // type=2 ("image")
    i32x8 g1;
    g1[0] = (int)((1u << 16)        // data_size = 1 -> 2 bytes
                | (1u << 20)        // pad_enable
                | (7u << 22)        // pad_interval code 7 -> every 256 DWORDs
                | (3u << 25));      // pad_amount code 3 -> 4 DWORDs (16B)
    g1[1] = (int)(512u << 16);      // tensor_dim0[15:0]=512 (atomic_bar_addr=0)
    g1[2] = (int)(16u  << 16);      // tensor_dim0[31:16]=0 | tensor_dim1[15:0]=16
    g1[3] = (int)(512u << 16);      // tensor_dim1[31:16]=0 | tile_dim0=512
    g1[4] = 16;                     // tile_dim1=16, tile_dim2=0
    g1[5] = 512;                    // tensor_dim0_stride[31:0]=512 elements
    g1[6] = 0;                      // stride hi / dim1_stride lo
    g1[7] = 0;
    i32x4 gz4 = {0, 0, 0, 0};       // groups 2/3 unused (2D tensor)
    i32x8 gz8 = {0, 0, 0, 0, 0, 0, 0, 0};
    // 6-arg form (amdgpu-toolchain / clang-23 signature)
    __builtin_amdgcn_tensor_load_to_lds(g0, g1, gz4, gz4, gz8, 0);
}

// ---------------------------------------------------------------------------
// Kernel 3: fused GEMM (zn @ zn^T) + streaming logsumexp via WMMA f16->f32.
// 64 blocks x 8 waves; wave w holds its 16x512 A-operand resident in VGPRs.
// B tiles (16x512 f16) are double-buffered in LDS via the Tensor Data Mover:
// wave 0 issues tensor_load_to_lds for tile j+1, all waves compute tile j
// (16 chained v_wmma_f32_16x16x32_f16 + one v_exp per element), then
// s_wait_tensorcnt + barrier publishes the next buffer. Max bound 1/T is
// known analytically, so no online max is needed.
// ---------------------------------------------------------------------------
__global__ __launch_bounds__(256) void ntx_lse(const _Float16* __restrict__ znh,
                                               float* __restrict__ lse) {
    __shared__ _Float16 Bs[2 * 16 * BSTRIDE];   // ~32.5 KB, double buffered
    constexpr unsigned TILE_LDS_BYTES = 16u * BSTRIDE * sizeof(_Float16);

    const int lane = threadIdx.x & 31;
    const int wave = threadIdx.x >> 5;          // 0..7
    const int g    = lane >> 4;                 // lane-group (K-half select)
    const int lm   = lane & 15;
    const int rowBase = blockIdx.x * 128 + wave * 16;

    // The TDM builtin only receives an integer LDS offset, so the compiler
    // sees no writer of Bs and would fold the ds_loads to undef. Escape the
    // buffer and declare that outside agents (the TDM) may write memory.
    asm volatile("" :: "v"(&Bs[0]) : "memory");

    // ---- Preload resident A fragments (CDNA5 16-bit A 16x32 layout):
    // lane group g holds K = k0 + 8g + e + (e>=8 ? 8 : 0)  for e=0..15
    v16h A[16];
    {
        const _Float16* arow = znh + (size_t)(rowBase + lm) * DDIM;
#pragma unroll
        for (int kc = 0; kc < 16; ++kc) {
            const int k0 = kc * 32 + g * 8;
            v8h lo = *(const v8h*)(arow + k0);
            v8h hi = *(const v8h*)(arow + k0 + 16);
            A[kc] = __builtin_shufflevector(lo, hi, 0, 1, 2, 3, 4, 5, 6, 7,
                                            8, 9, 10, 11, 12, 13, 14, 15);
        }
    }

    float s[8];
#pragma unroll
    for (int v = 0; v < 8; ++v) s[v] = 0.0f;

    const float kScale = INV_T * LOG2E;         // fold 1/T and log2(e)
    const float kBias  = INV_T * LOG2E;         // subtract bound*log2(e)

    // ---- prime buffer 0 with tile 0 (wave 0 drives the TDM)
    if (wave == 0) {
        tdm_load_tile(znh, 0);
        __builtin_amdgcn_s_wait_tensorcnt(0);
    }
    __syncthreads();

    for (int j = 0; j < 512; ++j) {
        // The freshly published LDS buffer was written by the TDM, which the
        // compiler cannot see: force B fragments to be (re)loaded this iter.
        asm volatile("" ::: "memory");

        const int colBase = j * 16;
        const _Float16* Bbuf = &Bs[(j & 1) * (16 * BSTRIDE)];

        // async-fetch next tile into the other buffer while we compute
        if (wave == 0 && (j + 1) < 512)
            tdm_load_tile(znh + (size_t)(colBase + 16) * DDIM,
                          (unsigned)((j + 1) & 1) * TILE_LDS_BYTES);

        // ---- 16x16 tile accumulate over K=512 (16 WMMAs)
        v8f c = {};
#pragma unroll
        for (int kc = 0; kc < 16; ++kc) {
            // B 32x16 layout: lane holds col lm, K = kc*32 + 8g + e + (e>=8 ? 8 : 0), e=0..15
            const _Float16* bp = Bbuf + lm * BSTRIDE + kc * 32 + g * 8;
            v8h blo = *(const v8h*)bp;
            v8h bhi = *(const v8h*)(bp + 16);
            v16h b = __builtin_shufflevector(blo, bhi, 0, 1, 2, 3, 4, 5, 6, 7,
                                             8, 9, 10, 11, 12, 13, 14, 15);
            c = __builtin_amdgcn_wmma_f32_16x16x32_f16(
                    false, A[kc], false, b, (short)0, c, false, false);
        }

        // ---- streaming exp-sum; C layout: c[v] <-> row v+8g, col lm
        const int gcol = colBase + lm;
#pragma unroll
        for (int v = 0; v < 8; ++v) {
            const int grow = rowBase + v + 8 * g;
            float e = __builtin_amdgcn_exp2f(c[v] * kScale - kBias);
            s[v] += (grow == gcol) ? 0.0f : e;   // diagonal -> exp(NEG_INF)=0
        }

        // ---- publish next buffer: TDM must be done, then workgroup barrier
        if (wave == 0 && (j + 1) < 512)
            __builtin_amdgcn_s_wait_tensorcnt(0);
        __syncthreads();
    }

    // ---- merge the 16 lane-slices of each row, emit lse = 1/T + ln(sum)
#pragma unroll
    for (int v = 0; v < 8; ++v) {
        float t = s[v];
        t += __shfl_xor(t, 1, 32);
        t += __shfl_xor(t, 2, 32);
        t += __shfl_xor(t, 4, 32);
        t += __shfl_xor(t, 8, 32);
        if (lm == 0)
            lse[rowBase + v + 8 * g] = INV_T + __builtin_amdgcn_logf(t) * LN2;
    }
}

// ---------------------------------------------------------------------------
// Kernel 4: mean over 8192 rows of (lse - pos)
// ---------------------------------------------------------------------------
__global__ __launch_bounds__(256) void ntx_reduce(const float* __restrict__ lse,
                                                  const float* __restrict__ pos,
                                                  float* __restrict__ out) {
    __shared__ float sm[256];
    float acc = 0.0f;
    for (int i = threadIdx.x; i < TWO_N; i += 256) acc += lse[i] - pos[i];
    sm[threadIdx.x] = acc;
    __syncthreads();
    for (int st = 128; st > 0; st >>= 1) {
        if (threadIdx.x < st) sm[threadIdx.x] += sm[threadIdx.x + st];
        __syncthreads();
    }
    if (threadIdx.x == 0) out[0] = sm[0] / (float)TWO_N;
}

// ---------------------------------------------------------------------------
extern "C" void kernel_launch(void* const* d_in, const int* in_sizes, int n_in,
                              void* d_out, int out_size, void* d_ws, size_t ws_size,
                              hipStream_t stream) {
    const float* z1 = (const float*)d_in[0];
    const float* z2 = (const float*)d_in[1];
    float* out = (float*)d_out;

    // workspace: [ znh f16 8192x512 | norms 8192 | pos 8192 | lse 8192 ]
    _Float16* znh = (_Float16*)d_ws;
    float* norms = (float*)((char*)d_ws + (size_t)TWO_N * DDIM * sizeof(_Float16));
    float* pos   = norms + TWO_N;
    float* lse   = pos + TWO_N;

    ntx_normalize<<<TWO_N / 8, 256, 0, stream>>>(z1, z2, znh, norms);
    ntx_positives<<<NHALF / 8, 256, 0, stream>>>(z1, z2, norms, pos);
    ntx_lse<<<TWO_N / 128, 256, 0, stream>>>(znh, lse);
    ntx_reduce<<<1, 256, 0, stream>>>(lse, pos, out);
}
